// SimpleLSTM_88390426952198
// MI455X (gfx1250) — compile-verified, matrix operand order checked by reading, not run on hardware
//
#include <hip/hip_runtime.h>
#include <stdint.h>

#define Bsz  32
#define Lseq 512
#define Din  1024
#define Hd   1024
#define G4   (4 * Hd)   // 4096 gate columns

typedef __attribute__((ext_vector_type(16))) __bf16 v16bf;
typedef __attribute__((ext_vector_type(8)))  float  v8f;
typedef __attribute__((ext_vector_type(4)))  int    v4i;

struct U128 { unsigned int w[4]; };
union FragB16 { U128 u[2]; v16bf v; };   // 32 B = 16 bf16 = one WMMA A/B fragment per lane

// ---- CDNA5 async global->LDS path (ASYNCcnt), guarded so compile never regresses ----
#if __has_builtin(__builtin_amdgcn_global_load_async_to_lds_b128)
#define HAVE_ASYNC_LDS 1
// pointer types must carry the address space via typedef (attribute in a cast is dropped)
typedef v4i __attribute__((address_space(1)))* gbl_v4i_p;   // printed as "__device__ *"
typedef v4i __attribute__((address_space(3)))* lds_v4i_p;   // printed as "__shared__ *"
#else
#define HAVE_ASYNC_LDS 0
#endif

__device__ __forceinline__ void async_cp_b128(void* lds_dst, const void* gsrc) {
#if HAVE_ASYNC_LDS
    __builtin_amdgcn_global_load_async_to_lds_b128(
        (gbl_v4i_p)gsrc, (lds_v4i_p)lds_dst, 0, 0);
#else
    *(U128*)lds_dst = *(const U128*)gsrc;
#endif
}

__device__ __forceinline__ void async_wait_all() {
#if HAVE_ASYNC_LDS
#if __has_builtin(__builtin_amdgcn_s_wait_asynccnt)
    __builtin_amdgcn_s_wait_asynccnt(0);
#else
    asm volatile("s_wait_asynccnt 0" ::: "memory");
#endif
#endif
}

__device__ __forceinline__ unsigned short f2bf(float f) {
    unsigned u = __float_as_uint(f);
    unsigned r = u + 0x7FFFu + ((u >> 16) & 1u);   // round-to-nearest-even
    return (unsigned short)(r >> 16);
}

__device__ __forceinline__ float sigf(float x) { return 1.0f / (1.0f + __expf(-x)); }

// ---------------------------------------------------------------------------
// 1) Convert x (B,L,D) f32 -> time-major bf16 xt (L,B,D)
// ---------------------------------------------------------------------------
__global__ void k_cvt_x(const float* __restrict__ x, unsigned short* __restrict__ xt) {
    size_t idx = (size_t)blockIdx.x * blockDim.x + threadIdx.x;   // over L*B*D
    if (idx >= (size_t)Lseq * Bsz * Din) return;
    size_t t = idx / (Bsz * Din);
    size_t rem = idx % (Bsz * Din);
    size_t b = rem / Din;
    size_t d = rem % Din;
    xt[idx] = f2bf(x[(b * Lseq + t) * Din + d]);
}

// ---------------------------------------------------------------------------
// 2) Split W (4H, H+D) f32 into bf16 Wh (4H,H) and Wx (4H,D)
// ---------------------------------------------------------------------------
__global__ void k_cvt_w(const float* __restrict__ W,
                        unsigned short* __restrict__ wh,
                        unsigned short* __restrict__ wx) {
    size_t idx = (size_t)blockIdx.x * blockDim.x + threadIdx.x;   // over 4H*(H+D)
    if (idx >= (size_t)G4 * (Hd + Din)) return;
    size_t n = idx / (Hd + Din);
    size_t k = idx % (Hd + Din);
    unsigned short v = f2bf(W[idx]);
    if (k < Hd) wh[n * Hd + k] = v;
    else        wx[n * Din + (k - Hd)] = v;
}

// ---------------------------------------------------------------------------
// 3) Zero h0 double-buffer + grid-barrier counters (re-run every call)
// ---------------------------------------------------------------------------
__global__ void k_init(unsigned short* __restrict__ hb, unsigned* __restrict__ bar) {
    int idx = blockIdx.x * blockDim.x + threadIdx.x;
    if (idx < 2 * Bsz * Hd) hb[idx] = 0;
    if (idx < 2) bar[idx] = 0;
}

// ---------------------------------------------------------------------------
// 4) Gx = Xt @ Wx^T + bias : (16384 x 1024) * (1024 x 4096) bf16 WMMA GEMM.
//    Block = 128x64 output tile, 8 waves x {1 M-tile, 4 N-tiles}.
//    Per 32-deep K chunk: A (8 KB) + B (4 KB) async-copied into LDS,
//    double-buffered so chunk k+1 streams in under chunk k's WMMAs.
// ---------------------------------------------------------------------------
__global__ void k_gx(const unsigned short* __restrict__ xt,
                     const unsigned short* __restrict__ wx,
                     const float* __restrict__ bias,
                     float* __restrict__ gx) {
    __shared__ unsigned short alds[2][128][32];   // 16 KB
    __shared__ unsigned short blds[2][64][32];    //  8 KB

    const int tid  = threadIdx.x;
    const int lane = tid & 31;
    const int wid  = tid >> 5;
    const int mb   = blockIdx.x >> 6;   // 0..127 : 128-row slab
    const int nb   = blockIdx.x & 63;   // 0..63  : 64-col slab
    const int m    = lane & 15;
    const int hi   = lane >> 4;

    auto stage = [&](int buf, int kk) {
        // A: 128 rows x 32 bf16 = 512 U128 chunks -> 2 per thread
        for (int i = tid; i < 512; i += 256) {
            const int r = i >> 2, q = i & 3;
            async_cp_b128(&alds[buf][r][q * 8],
                          xt + (size_t)(mb * 128 + r) * Din + kk * 32 + q * 8);
        }
        // B: 64 rows x 32 bf16 = 256 U128 chunks -> 1 per thread
        {
            const int r = tid >> 2, q = tid & 3;
            async_cp_b128(&blds[buf][r][q * 8],
                          wx + (size_t)(nb * 64 + r) * Din + kk * 32 + q * 8);
        }
    };

    stage(0, 0);

    v8f acc[4] = {{}, {}, {}, {}};
    for (int kk = 0; kk < Din / 32; ++kk) {
        const int buf = kk & 1;
        async_wait_all();
        __syncthreads();                         // staged chunk kk visible to all
        if (kk + 1 < Din / 32) stage(buf ^ 1, kk + 1);

        const unsigned short* arow = &alds[buf][wid * 16 + m][0];
        FragB16 a;
        a.u[0] = *(const U128*)(arow + hi * 8);          // ds_load_b128
        a.u[1] = *(const U128*)(arow + hi * 8 + 16);
#pragma unroll
        for (int s = 0; s < 4; ++s) {
            const unsigned short* brow = &blds[buf][s * 16 + m][0];
            FragB16 b;
            b.u[0] = *(const U128*)(brow + hi * 8);
            b.u[1] = *(const U128*)(brow + hi * 8 + 16);
            acc[s] = __builtin_amdgcn_wmma_f32_16x16x32_bf16(
                false, a.v, false, b.v, (short)0, acc[s], false, false);
        }
    }

#pragma unroll
    for (int s = 0; s < 4; ++s) {
        const int col = nb * 64 + s * 16 + (lane & 15);
        const float bb = bias[col];
#pragma unroll
        for (int r = 0; r < 8; ++r) {
            const int row = mb * 128 + wid * 16 + hi * 8 + r;  // C/D: VGPR r -> M=r / r+8
            gx[(size_t)row * G4 + col] = acc[s][r] + bb;
        }
    }
}

// ---------------------------------------------------------------------------
// 5) Persistent recurrent kernel. 64 blocks x 256 threads (8 waves).
//    Block blk owns h columns [blk*16, blk*16+16).
//    LDS: Wh rows (4 gates x 16 cols x 1024 K, 128 KB, staged once) +
//         h_t broadcast (32x1024 bf16, 64 KB, staged per step) +
//         gate exchange (8 KB)  => 200 KB of the 320 KB WGP LDS.
//    Wave w: gate g=w>>1, batch-half mt=w&1 -> one 16x16 WMMA tile/step,
//    two independent accumulator chains over even/odd K chunks.
//    c lives in registers; h published bf16 via L2 + atomic grid barrier.
// ---------------------------------------------------------------------------
__global__ void k_lstm(const unsigned short* __restrict__ wh,
                       const float* __restrict__ gx,
                       unsigned short* __restrict__ hbuf,   // 2 * 32 * 1024 bf16
                       float* __restrict__ out,             // (B, L, H) f32
                       unsigned* __restrict__ bar) {
    extern __shared__ char smem[];
    unsigned short (*wlds)[Hd] = (unsigned short(*)[Hd])smem;               // [64][1024]
    unsigned short (*hlds)[Hd] = (unsigned short(*)[Hd])(smem + 131072);    // [32][1024]
    float (*gbuf)[Bsz][16]     = (float(*)[Bsz][16])(smem + 131072 + 65536);// [4][32][16]

    const int tid  = threadIdx.x;
    const int lane = tid & 31;
    const int wid  = tid >> 5;
    const int g    = wid >> 1;          // gate 0..3 (f,i,o,c~)
    const int mt   = wid & 1;           // batch half
    const int mbase = mt * 16;
    const int m    = lane & 15;
    const int hi   = lane >> 4;
    const int blk  = blockIdx.x;        // 0..63
    const int n0   = g * Hd + blk * 16; // gate-column base in 4096

    // ---- stage this block's 64 Wh rows into LDS once (async, 128 KB) ----
    for (int i = tid; i < 64 * (Hd / 8); i += 256) {      // 8192 U128 chunks
        const int r = i >> 7, kq = i & 127;
        const int gg = r >> 4, cc = r & 15;
        async_cp_b128(((U128*)wlds[r]) + kq,
                      (const U128*)(wh + (size_t)(gg * Hd + blk * 16 + cc) * Hd) + kq);
    }
    async_wait_all();
    __syncthreads();

    const unsigned short* browE = wlds[g * 16 + m];       // B fragments (LDS resident)

    float c0 = 0.0f, c1 = 0.0f;                           // persistent cell state
    const int e0 = tid, e1 = tid + 256;                   // 512 elems: row=e>>4, col=e&15

    for (int t = 0; t < Lseq; ++t) {
        // ---- broadcast h_{t-1} into LDS (async, 64 KB, shared by 8 waves) ----
        const unsigned short* hcur = hbuf + (size_t)(t & 1) * (Bsz * Hd);
        for (int i = tid; i < Bsz * (Hd / 8); i += 256) { // 4096 U128 chunks
            const int r = i >> 7, kq = i & 127;
            async_cp_b128(((U128*)hlds[r]) + kq,
                          (const U128*)(hcur + (size_t)r * Hd) + kq);
        }
        async_wait_all();
        __syncthreads();

        const unsigned short* arow = hlds[mbase + m];

        // ---- 16x16 gate tile: K=1024, two independent WMMA chains ----
        v8f accE = {}, accO = {};
#pragma unroll 8
        for (int kk = 0; kk < Hd / 32; kk += 2) {
            const int k0 = kk * 32 + hi * 8;
            FragB16 a0, b0, a1, b1;
            a0.u[0] = *(const U128*)(arow + k0);
            a0.u[1] = *(const U128*)(arow + k0 + 16);
            b0.u[0] = *(const U128*)(browE + k0);
            b0.u[1] = *(const U128*)(browE + k0 + 16);
            a1.u[0] = *(const U128*)(arow + k0 + 32);
            a1.u[1] = *(const U128*)(arow + k0 + 48);
            b1.u[0] = *(const U128*)(browE + k0 + 32);
            b1.u[1] = *(const U128*)(browE + k0 + 48);
            accE = __builtin_amdgcn_wmma_f32_16x16x32_bf16(
                false, a0.v, false, b0.v, (short)0, accE, false, false);
            accO = __builtin_amdgcn_wmma_f32_16x16x32_bf16(
                false, a1.v, false, b1.v, (short)0, accO, false, false);
        }

        // ---- add precomputed x-contribution (bias folded in) ----
        const int col16 = lane & 15;
        const size_t gxbase = (size_t)t * Bsz * G4;
#pragma unroll
        for (int r = 0; r < 8; ++r) {
            const int row = mbase + hi * 8 + r;
            gbuf[g][row][col16] = accE[r] + accO[r]
                                + gx[gxbase + (size_t)row * G4 + n0 + col16];
        }
        if (t + 1 < Lseq)   // pull next step's Gx slice toward the WGP
            __builtin_prefetch(gx + gxbase + (size_t)Bsz * G4
                               + (size_t)(mbase + hi * 8) * G4 + n0 + col16, 0, 1);
        __syncthreads();

        // ---- elementwise gate math: each thread owns 2 of 512 cells ----
        unsigned short* hnext = hbuf + (size_t)((t + 1) & 1) * (Bsz * Hd);
        {
            const int row = e0 >> 4, col = e0 & 15;
            const float f = gbuf[0][row][col], i = gbuf[1][row][col];
            const float o = gbuf[2][row][col], ct = gbuf[3][row][col];
            c0 = sigf(f) * c0 + sigf(i) * tanhf(ct);
            const float h = sigf(o) * tanhf(c0);
            out[(size_t)row * (Lseq * Hd) + (size_t)t * Hd + blk * 16 + col] = h;
            hnext[row * Hd + blk * 16 + col] = f2bf(h);
        }
        {
            const int row = e1 >> 4, col = e1 & 15;
            const float f = gbuf[0][row][col], i = gbuf[1][row][col];
            const float o = gbuf[2][row][col], ct = gbuf[3][row][col];
            c1 = sigf(f) * c1 + sigf(i) * tanhf(ct);
            const float h = sigf(o) * tanhf(c1);
            out[(size_t)row * (Lseq * Hd) + (size_t)t * Hd + blk * 16 + col] = h;
            hnext[row * Hd + blk * 16 + col] = f2bf(h);
        }

        // ---- grid barrier: all 64 blocks see full h_t before step t+1 ----
        if (tid == 0) {
            __threadfence();
            const unsigned gen = __hip_atomic_load(bar + 1, __ATOMIC_RELAXED,
                                                   __HIP_MEMORY_SCOPE_AGENT);
            const unsigned arr = __hip_atomic_fetch_add(bar + 0, 1u, __ATOMIC_ACQ_REL,
                                                        __HIP_MEMORY_SCOPE_AGENT);
            if (arr == gridDim.x - 1) {
                __hip_atomic_store(bar + 0, 0u, __ATOMIC_RELAXED,
                                   __HIP_MEMORY_SCOPE_AGENT);
                __hip_atomic_fetch_add(bar + 1, 1u, __ATOMIC_RELEASE,
                                       __HIP_MEMORY_SCOPE_AGENT);
            } else {
                while (__hip_atomic_load(bar + 1, __ATOMIC_ACQUIRE,
                                         __HIP_MEMORY_SCOPE_AGENT) == gen)
                    __builtin_amdgcn_s_sleep(2);
            }
        }
        __syncthreads();
    }
}

// ---------------------------------------------------------------------------
// Workspace layout (bytes):
//   xt   bf16[L*B*D]   @ 0          (33,554,432)
//   wh   bf16[4H*H]    @ 33554432   ( 8,388,608)
//   wx   bf16[4H*D]    @ 41943040   ( 8,388,608)
//   gx   f32 [L*B*4H]  @ 50331648   (268,435,456)
//   hbuf bf16[2*B*H]   @ 318767104  (   131,072)
//   bar  u32 [2]       @ 318898176
// ---------------------------------------------------------------------------
extern "C" void kernel_launch(void* const* d_in, const int* in_sizes, int n_in,
                              void* d_out, int out_size, void* d_ws, size_t ws_size,
                              hipStream_t stream) {
    const float* x    = (const float*)d_in[0];
    const float* W    = (const float*)d_in[1];
    const float* bias = (const float*)d_in[2];
    float* out = (float*)d_out;

    char* ws = (char*)d_ws;
    unsigned short* xt = (unsigned short*)(ws);
    unsigned short* wh = (unsigned short*)(ws + 33554432);
    unsigned short* wx = (unsigned short*)(ws + 41943040);
    float*          gx = (float*)(ws + 50331648);
    unsigned short* hb = (unsigned short*)(ws + 318767104);
    unsigned*       br = (unsigned*)(ws + 318898176);

    k_cvt_x<<<65536, 256, 0, stream>>>(x, xt);
    k_cvt_w<<<32768, 256, 0, stream>>>(W, wh, wx);
    k_init <<<256,   256, 0, stream>>>(hb, br);
    k_gx   <<<8192,  256, 0, stream>>>(xt, wx, bias, gx);

    // 128 KB Wh + 64 KB h + 8 KB gates = 204,800 B dynamic LDS (< 320 KB WGP)
    const size_t smem = 131072 + 65536 + 8192;
    k_lstm <<<64, 256, smem, stream>>>(wh, gx, hb, out, br);
}
